// RelationGraphAttention_65000035058007
// MI455X (gfx1250) — compile-verified, matrix operand order checked by reading, not run on hardware
//
#include <hip/hip_runtime.h>
#include <hip/hip_bf16.h>
#include <stdint.h>

typedef __attribute__((ext_vector_type(16))) __bf16 v16bf;
typedef __attribute__((ext_vector_type(8)))  float  v8f;

#define H   128
#define ED  32
#define LNEPS 1e-5f

// ---------------------------------------------------------------------------
// utility
// ---------------------------------------------------------------------------
__global__ void zero_f32_kernel(float* __restrict__ p, int n) {
  int i = blockIdx.x * blockDim.x + threadIdx.x;
  if (i < n) p[i] = 0.f;
}

// order-preserving float<->uint encoding for atomicMax-based segment max
__device__ __forceinline__ unsigned enc_f32(float x) {
  unsigned u = __float_as_uint(x);
  return (u & 0x80000000u) ? ~u : (u | 0x80000000u);
}
__device__ __forceinline__ float dec_f32(unsigned u) {
  return __uint_as_float((u & 0x80000000u) ? (u & 0x7FFFFFFFu) : ~u);
}

// pack 4 f32 -> 4 bf16, single 8-byte LDS store
__device__ __forceinline__ void store4_bf16(__bf16* dst, float a, float b,
                                            float c, float d) {
  union { __bf16 h[4]; uint2 u; } pk;
  pk.h[0] = (__bf16)a; pk.h[1] = (__bf16)b;
  pk.h[2] = (__bf16)c; pk.h[3] = (__bf16)d;
  *(uint2*)dst = pk.u;
}

// ---------------------------------------------------------------------------
// bf16 WMMA GEMM:  C[M,N] = A[M,K] * B[K,N] (+bias[N]) (+Cin[M,N])
// f32 in memory, converted to bf16 while staging into LDS (vectorized, packed
// ds_store_b64), register-buffered pipeline so global loads of tile i+1
// overlap WMMA compute of tile i. OOB rows are pointer-clamped to row 0 and
// zeroed with a branchless select (no exec-mask divergence in the load path).
// N multiple of 64, K multiple of 32 (and of 4 for float4 alignment).
// Block = 128 threads = 4 waves (wave32); block tile 64x64.
// ---------------------------------------------------------------------------
#define BM 64
#define BN 64
#define BK 32
#define AS_LD 40
#define BS_LD 40

__global__ __launch_bounds__(128)
void gemm_bf16_wmma(const float* __restrict__ A, const float* __restrict__ B,
                    const float* __restrict__ bias, const float* __restrict__ Cin,
                    float* __restrict__ C, int M, int N, int K) {
  __shared__ __bf16 As [BM * AS_LD];      // [row][k]
  __shared__ __bf16 Bst[BN * BS_LD];      // transposed: [n][k]

  const int tid  = threadIdx.x;
  const int wave = tid >> 5;
  const int lane = tid & 31;
  const int half = lane >> 4;             // 0: lanes 0-15, 1: lanes 16-31
  const int l16  = lane & 15;

  const int row0 = blockIdx.y * BM;
  const int col0 = blockIdx.x * BN;

  // ---- A staging plan: thread owns 4 float4 groups of the 64x32 tile ----
  const float* aPtr[4];
  __bf16*      aLds[4];
  float        aMask[4];                  // 1.0 in-range, 0.0 out-of-range
  #pragma unroll
  for (int c = 0; c < 4; ++c) {
    int g  = tid + c * 128;               // group id, 512 groups of 4 floats
    int r  = g >> 3;                      // row in tile (8 groups per row)
    int kc = (g & 7) * 4;                 // k offset, multiple of 4
    int gr = row0 + r;
    bool ok = (gr < M);
    aMask[c] = ok ? 1.f : 0.f;
    aPtr[c] = A + (size_t)(ok ? gr : 0) * K + kc;   // always a valid address
    aLds[c] = &As[r * AS_LD + kc];
  }
  // ---- B staging plan: thread owns column n, k-quads {kq*4 + p*8 + j} ----
  const int nB  = tid & 63;
  const int kq4 = (tid >> 6) * 4;
  const float* bCur = B + col0 + nB + (size_t)kq4 * N;
  __bf16* bLds = &Bst[nB * BS_LD + kq4];

  float4 aR[4];
  float  bR[4][4];

  auto loadTile = [&]() {
    #pragma unroll
    for (int c = 0; c < 4; ++c) {
      float4 v = *(const float4*)aPtr[c];           // unconditional, clamped
      v.x *= aMask[c]; v.y *= aMask[c];             // branchless zero of OOB
      v.z *= aMask[c]; v.w *= aMask[c];
      aR[c] = v;
    }
    #pragma unroll
    for (int p = 0; p < 4; ++p)
      #pragma unroll
      for (int j = 0; j < 4; ++j)
        bR[p][j] = bCur[(size_t)(p * 8 + j) * N];
  };

  v8f acc[4] = {};                        // 4 C tiles of 16x16 f32
  loadTile();                             // preload tile 0

  for (int k0 = 0; k0 < K; k0 += BK) {
    // convert + store staged registers into LDS (packed b64 stores)
    #pragma unroll
    for (int c = 0; c < 4; ++c)
      store4_bf16(aLds[c], aR[c].x, aR[c].y, aR[c].z, aR[c].w);
    #pragma unroll
    for (int p = 0; p < 4; ++p)
      store4_bf16(bLds + p * 8, bR[p][0], bR[p][1], bR[p][2], bR[p][3]);
    __syncthreads();

    // kick off next tile's global loads; they complete while we do WMMAs
    if (k0 + BK < K) {
      #pragma unroll
      for (int c = 0; c < 4; ++c) aPtr[c] += BK;
      bCur += (size_t)BK * N;
      loadTile();
    }

    // A fragment: lane holds row l16 of this wave's strip; per ISA layout,
    // elements 0..7 = K 8*half.., elements 8..15 = K 16+8*half..
    v16bf af;
    {
      const __bf16* ap = &As[(wave * 16 + l16) * AS_LD + 8 * half];
      #pragma unroll
      for (int j = 0; j < 8; ++j) af[j] = ap[j];
      #pragma unroll
      for (int j = 0; j < 8; ++j) af[8 + j] = ap[16 + j];
    }
    #pragma unroll
    for (int t = 0; t < 4; ++t) {
      v16bf bfr;                          // B fragment, mirrored layout on [n][k]
      const __bf16* bp = &Bst[(t * 16 + l16) * BS_LD + 8 * half];
      #pragma unroll
      for (int j = 0; j < 8; ++j) bfr[j] = bp[j];
      #pragma unroll
      for (int j = 0; j < 8; ++j) bfr[8 + j] = bp[16 + j];
      acc[t] = __builtin_amdgcn_wmma_f32_16x16x32_bf16(
          false, af, false, bfr, (short)0, acc[t], false, false);
    }
    __syncthreads();
  }

  // C layout: VGPR r -> row r + 8*half of the tile, col = l16
  #pragma unroll
  for (int t = 0; t < 4; ++t) {
    int col = col0 + t * 16 + l16;
    #pragma unroll
    for (int r = 0; r < 8; ++r) {
      int grow = row0 + wave * 16 + r + 8 * half;
      if (grow < M) {
        float v = acc[t][r];
        if (bias) v += bias[col];
        if (Cin)  v += Cin[(size_t)grow * N + col];
        C[(size_t)grow * N + col] = v;
      }
    }
  }
}

// ---------------------------------------------------------------------------
// bem[n] = bmsg[n] + sum_k be[k] * Wmsg[k][n]       (1 block, 128 threads)
// ---------------------------------------------------------------------------
__global__ __launch_bounds__(128)
void bem_kernel(const float* __restrict__ be, const float* __restrict__ Wmsg,
                const float* __restrict__ bmsg, float* __restrict__ bem) {
  int n = threadIdx.x;
  float acc = bmsg[n];
  for (int k = 0; k < H; ++k) acc += be[k] * Wmsg[k * H + n];
  bem[n] = acc;
}

// ---------------------------------------------------------------------------
// wave-per-edge attention scores + segment max (atomicMax on ordered uint).
// Lane q owns H components n = 4*lane..4*lane+3 -> all row reads are b128.
// ---------------------------------------------------------------------------
__global__ __launch_bounds__(256)
void edge_score_kernel(const long long* __restrict__ src_idx,
                       const long long* __restrict__ dst_idx,
                       const float* __restrict__ edge_attr,
                       const float* __restrict__ S, const float* __restrict__ D,
                       const float* __restrict__ We, const float* __restrict__ be,
                       const float* __restrict__ Wattn, const float* __restrict__ battn,
                       float* __restrict__ scores, unsigned* __restrict__ smax, int E) {
  int e    = (blockIdx.x * blockDim.x + threadIdx.x) >> 5;
  int lane = threadIdx.x & 31;
  if (e >= E) return;
  long long s = src_idx[e], d = dst_idx[e];
  float ea = edge_attr[(size_t)e * ED + lane];    // lane k holds edge_attr[e][k]

  const float4* We4 = (const float4*)We;          // [k][H/4]
  float4 ef = ((const float4*)be)[lane];
  #pragma unroll
  for (int k = 0; k < ED; ++k) {
    float eak = __shfl(ea, k, 32);
    float4 w = We4[k * (H / 4) + lane];
    ef.x += eak * w.x; ef.y += eak * w.y;
    ef.z += eak * w.z; ef.w += eak * w.w;
  }
  float4 sf  = ((const float4*)(S + (size_t)s * H))[lane];
  float4 df  = ((const float4*)(D + (size_t)d * H))[lane];
  float4 wa0 = ((const float4*)(Wattn))[lane];
  float4 wa1 = ((const float4*)(Wattn + H))[lane];
  float4 wa2 = ((const float4*)(Wattn + 2 * H))[lane];
  float sc = tanhf(sf.x) * wa0.x + tanhf(sf.y) * wa0.y +
             tanhf(sf.z) * wa0.z + tanhf(sf.w) * wa0.w +
             tanhf(df.x) * wa1.x + tanhf(df.y) * wa1.y +
             tanhf(df.z) * wa1.z + tanhf(df.w) * wa1.w +
             tanhf(ef.x) * wa2.x + tanhf(ef.y) * wa2.y +
             tanhf(ef.z) * wa2.z + tanhf(ef.w) * wa2.w;
  #pragma unroll
  for (int off = 16; off > 0; off >>= 1) sc += __shfl_xor(sc, off, 32);
  if (lane == 0) {
    float sco = sc + battn[0];
    scores[e] = sco;
    atomicMax(&smax[d], enc_f32(sco));
  }
}

// ---------------------------------------------------------------------------
// ex = exp(score - segmax[dst]);  segment sum via atomicAdd
// ---------------------------------------------------------------------------
__global__ __launch_bounds__(256)
void edge_exp_kernel(const long long* __restrict__ dst_idx,
                     const float* __restrict__ scores,
                     const unsigned* __restrict__ smax,
                     float* __restrict__ ex, float* __restrict__ ssum, int E) {
  int e = blockIdx.x * blockDim.x + threadIdx.x;
  if (e >= E) return;
  long long d = dst_idx[e];
  float x = __expf(scores[e] - dec_f32(smax[d]));
  ex[e] = x;
  atomicAdd(&ssum[d], x);
}

// ---------------------------------------------------------------------------
// wave-per-edge: msg = SM[src] + edge_attr @ Wem + bem; agg[dst] += alpha*msg
// ---------------------------------------------------------------------------
__global__ __launch_bounds__(256)
void edge_agg_kernel(const long long* __restrict__ src_idx,
                     const long long* __restrict__ dst_idx,
                     const float* __restrict__ edge_attr,
                     const float* __restrict__ SM, const float* __restrict__ Wem,
                     const float* __restrict__ bem, const float* __restrict__ ex,
                     const float* __restrict__ ssum, float* __restrict__ agg, int E) {
  int e    = (blockIdx.x * blockDim.x + threadIdx.x) >> 5;
  int lane = threadIdx.x & 31;
  if (e >= E) return;
  long long s = src_idx[e], d = dst_idx[e];
  float alpha = ex[e] / ssum[d];
  float ea = edge_attr[(size_t)e * ED + lane];

  const float4* Wem4 = (const float4*)Wem;        // [k][H/4]
  float4 m  = ((const float4*)bem)[lane];
  float4 sm = ((const float4*)(SM + (size_t)s * H))[lane];
  m.x += sm.x; m.y += sm.y; m.z += sm.z; m.w += sm.w;
  #pragma unroll
  for (int k = 0; k < ED; ++k) {
    float eak = __shfl(ea, k, 32);
    float4 w = Wem4[k * (H / 4) + lane];
    m.x += eak * w.x; m.y += eak * w.y;
    m.z += eak * w.z; m.w += eak * w.w;
  }
  float* dst = agg + (size_t)d * H + lane * 4;
  atomicAdd(dst + 0, alpha * m.x);
  atomicAdd(dst + 1, alpha * m.y);
  atomicAdd(dst + 2, alpha * m.z);
  atomicAdd(dst + 3, alpha * m.w);
}

// ---------------------------------------------------------------------------
// h = dst_x + U; LayerNorm(h) * gamma + beta      (block of 128 per row)
// ---------------------------------------------------------------------------
__global__ __launch_bounds__(128)
void out_ln_kernel(const float* __restrict__ dst_x, const float* __restrict__ U,
                   const float* __restrict__ gamma, const float* __restrict__ beta,
                   float* __restrict__ out) {
  int row = blockIdx.x;
  int t   = threadIdx.x;
  float h = dst_x[(size_t)row * H + t] + U[(size_t)row * H + t];

  __shared__ float part[4];
  float v = h;
  #pragma unroll
  for (int off = 16; off > 0; off >>= 1) v += __shfl_xor(v, off, 32);
  if ((t & 31) == 0) part[t >> 5] = v;
  __syncthreads();
  float mu = (part[0] + part[1] + part[2] + part[3]) * (1.f / H);
  float dff = h - mu;
  float v2 = dff * dff;
  __syncthreads();
  #pragma unroll
  for (int off = 16; off > 0; off >>= 1) v2 += __shfl_xor(v2, off, 32);
  if ((t & 31) == 0) part[t >> 5] = v2;
  __syncthreads();
  float var = (part[0] + part[1] + part[2] + part[3]) * (1.f / H);
  out[(size_t)row * H + t] = dff * rsqrtf(var + LNEPS) * gamma[t] + beta[t];
}

// ---------------------------------------------------------------------------
extern "C" void kernel_launch(void* const* d_in, const int* in_sizes, int n_in,
                              void* d_out, int out_size, void* d_ws, size_t ws_size,
                              hipStream_t stream) {
  const float*     src_x  = (const float*)d_in[0];
  const float*     dst_x  = (const float*)d_in[1];
  const long long* eidx   = (const long long*)d_in[2];   // int64 [2,E]
  const float*     eattr  = (const float*)d_in[3];
  const float*     Wsrc   = (const float*)d_in[4];
  const float*     bsrc   = (const float*)d_in[5];
  const float*     Wdst   = (const float*)d_in[6];
  const float*     bdst   = (const float*)d_in[7];
  const float*     We     = (const float*)d_in[8];
  const float*     be     = (const float*)d_in[9];
  const float*     Wattn  = (const float*)d_in[10];
  const float*     battn  = (const float*)d_in[11];
  const float*     Wmsg   = (const float*)d_in[12];
  const float*     bmsg   = (const float*)d_in[13];
  const float*     Wout   = (const float*)d_in[14];      // [2H, H]
  const float*     bout   = (const float*)d_in[15];
  const float*     gamma  = (const float*)d_in[16];
  const float*     beta   = (const float*)d_in[17];

  const int Nsrc = in_sizes[0] / H;
  const int Ndst = in_sizes[1] / H;
  const int E    = in_sizes[3] / ED;
  const long long* sidx = eidx;
  const long long* didx = eidx + E;

  // workspace layout (floats); smax/ssum/agg contiguous for single zero pass
  float* ws   = (float*)d_ws;
  float* S    = ws;                 ws += (size_t)Nsrc * H;
  float* D    = ws;                 ws += (size_t)Ndst * H;
  float* SM   = ws;                 ws += (size_t)Nsrc * H;
  float* Wem  = ws;                 ws += (size_t)ED * H;
  float* bem  = ws;                 ws += H;
  float* scr  = ws;                 ws += E;
  float* ex   = ws;                 ws += E;
  unsigned* smax = (unsigned*)ws;   ws += Ndst;           // zero == -inf sentinel
  float* ssum = ws;                 ws += Ndst;
  float* agg  = ws;                 ws += (size_t)Ndst * H;
  float* U    = ws;                 ws += (size_t)Ndst * H;
  float* outp = (float*)d_out;
  (void)n_in; (void)out_size; (void)ws_size;

  dim3 blk128(128), blk256(256);
  dim3 gS(H / BN, (Nsrc + BM - 1) / BM);
  dim3 gD(H / BN, (Ndst + BM - 1) / BM);
  dim3 gW(H / BN, (ED + BM - 1) / BM);

  // zero softmax / aggregation state (smax..agg is one contiguous f32 region)
  {
    int zn = Ndst + Ndst + Ndst * H;
    zero_f32_kernel<<<(zn + 255) / 256, blk256, 0, stream>>>((float*)smax, zn);
  }

  // node-level WMMA GEMMs
  gemm_bf16_wmma<<<gS, blk128, 0, stream>>>(src_x, Wsrc, bsrc, nullptr, S,  Nsrc, H, H);
  gemm_bf16_wmma<<<gD, blk128, 0, stream>>>(dst_x, Wdst, bdst, nullptr, D,  Ndst, H, H);
  gemm_bf16_wmma<<<gS, blk128, 0, stream>>>(S,     Wmsg, nullptr, nullptr, SM, Nsrc, H, H);
  gemm_bf16_wmma<<<gW, blk128, 0, stream>>>(We,    Wmsg, nullptr, nullptr, Wem, ED, H, H);
  bem_kernel<<<1, blk128, 0, stream>>>(be, Wmsg, bmsg, bem);

  // edge phase: scores + segment softmax + weighted aggregation
  int waveBlocks = (E + 7) / 8;   // 8 waves (256 threads) per block
  edge_score_kernel<<<waveBlocks, blk256, 0, stream>>>(
      sidx, didx, eattr, S, D, We, be, Wattn, battn, scr, smax, E);
  edge_exp_kernel<<<(E + 255) / 256, blk256, 0, stream>>>(didx, scr, smax, ex, ssum, E);
  edge_agg_kernel<<<waveBlocks, blk256, 0, stream>>>(
      sidx, didx, eattr, SM, Wem, bem, ex, ssum, agg, E);

  // updated = dst_x @ Wout[:H] + bout + agg @ Wout[H:]
  gemm_bf16_wmma<<<gD, blk128, 0, stream>>>(dst_x, Wout,          bout,    nullptr, U, Ndst, H, H);
  gemm_bf16_wmma<<<gD, blk128, 0, stream>>>(agg,   Wout + H * H,  nullptr, U,       U, Ndst, H, H);

  // residual + LayerNorm
  out_ln_kernel<<<Ndst, blk128, 0, stream>>>(dst_x, U, gamma, beta, outp);
}
